// TensorizedTransformerLM_50285477101989
// MI455X (gfx1250) — compile-verified, matrix operand order checked by reading, not run on hardware
//
#include <hip/hip_runtime.h>

// TensorizedTransformerLM for MI455X (gfx1250), wave32 + f16 WMMA (f32 accum).
constexpr int kS  = 256;   // seq len
constexpr int kB  = 2;     // batch
constexpr int kDM = 512;   // d_model
constexpr int kH  = 512;   // n_head*d_head
constexpr int kDI = 2048;  // d_inner
constexpr int kL  = 2;     // layers
constexpr float kEPS = 1e-5f;

typedef __attribute__((ext_vector_type(16))) _Float16 v16h;
typedef __attribute__((ext_vector_type(8)))  _Float16 v8h;
typedef __attribute__((ext_vector_type(8)))  float    v8f;

// ---------------------------------------------------------------------------
// Fragment layout (wave32, 16-bit A/B 16x32, CDNA5 ISA 7.12.2):
//   lane = half*16 + r   (r = row for A / col for B)
//   element e (0..15): k = ((e>>3)<<4) + (half<<3) + (e&7)
// => per lane, elements are two contiguous runs of 8:
//   e 0..7  <- [k0 + 8*half,      +8)
//   e 8..15 <- [k0 + 16 + 8*half, +8)
// C/D v8f: element e -> M = e + 8*half, N = lane&15
//   => per lane: fixed N, 8 consecutive M. Storing into an [N][M] layout
//      gives fully contiguous per-lane stores.
// ---------------------------------------------------------------------------

// Load 16 f32 (two contiguous runs of 8, second at +16) via 4 x b128.
__device__ __forceinline__ void load16f(const float* __restrict__ p, float* v) {
  const float4* q = (const float4*)p;
  float4 a = q[0], b = q[1], c = q[4], d = q[5];
  v[0] = a.x; v[1] = a.y; v[2]  = a.z; v[3]  = a.w;
  v[4] = b.x; v[5] = b.y; v[6]  = b.z; v[7]  = b.w;
  v[8] = c.x; v[9] = c.y; v[10] = c.z; v[11] = c.w;
  v[12] = d.x; v[13] = d.y; v[14] = d.z; v[15] = d.w;
}

__device__ __forceinline__ v16h load_frag_f32(const float* __restrict__ P, long ld,
                                              int r0, int k0) {
  int lane = threadIdx.x & 31;
  int half = lane >> 4;
  const float* p = P + (long)(r0 + (lane & 15)) * ld + k0 + half * 8;
  float v[16];
  load16f(p, v);
  v16h f;
#pragma unroll
  for (int e = 0; e < 16; ++e) f[e] = (_Float16)v[e];
  return f;
}

__device__ __forceinline__ v16h load_frag_f16(const _Float16* __restrict__ P, long ld,
                                              int r0, int k0) {
  int lane = threadIdx.x & 31;
  int half = lane >> 4;
  const _Float16* p = P + (long)(r0 + (lane & 15)) * ld + k0 + half * 8;
  v8h lo = *(const v8h*)(p);        // 16B
  v8h hi = *(const v8h*)(p + 16);   // 16B
  return __builtin_shufflevector(lo, hi, 0, 1, 2, 3, 4, 5, 6, 7,
                                 8, 9, 10, 11, 12, 13, 14, 15);
}

// ---------------------------------------------------------------------------
// Generic C = A[M,K] * B[N,K]^T  (A f32, B f32 or f16; f16 WMMA, f32 acc)
// TR_STORE=0: C row-major [M][N] (+bias[n], optional relu)
// TR_STORE=1: C stored as [N][M] with per-lane contiguous float4 stores
// Block: 256 threads = 8 waves, WG tile 128x128, wave tile 64x32.
// gridDim.z K-chunks write partial C slabs (deterministic split-K).
// ---------------------------------------------------------------------------
template <bool B_IS_F16, bool TR_STORE>
__global__ __launch_bounds__(256, 1) void
k_gemm_nt(const float* __restrict__ A, long lda,
          const float* __restrict__ Bf,
          const _Float16* __restrict__ Bh, long ldb,
          float* __restrict__ C, long ldc,
          long c_chunk_stride, int Kchunk,
          const float* __restrict__ bias, int relu) {
  int wv = threadIdx.x >> 5;
  int wm = wv >> 2;                       // 0..1
  int wn = wv & 3;                        // 0..3
  int m0 = blockIdx.x * 128 + wm * 64;
  int n0 = blockIdx.y * 128 + wn * 32;
  int k0 = blockIdx.z * Kchunk;
  C += (long)blockIdx.z * c_chunk_stride;

  v8f acc[4][2];
#pragma unroll
  for (int i = 0; i < 4; ++i)
#pragma unroll
    for (int j = 0; j < 2; ++j) acc[i][j] = {};

  for (int k = k0; k < k0 + Kchunk; k += 32) {
    v16h af[4];
#pragma unroll
    for (int i = 0; i < 4; ++i) af[i] = load_frag_f32(A, lda, m0 + 16 * i, k);
    v16h bf[2];
#pragma unroll
    for (int j = 0; j < 2; ++j) {
      if constexpr (B_IS_F16) bf[j] = load_frag_f16(Bh, ldb, n0 + 16 * j, k);
      else                    bf[j] = load_frag_f32(Bf, ldb, n0 + 16 * j, k);
    }
#pragma unroll
    for (int i = 0; i < 4; ++i)
#pragma unroll
      for (int j = 0; j < 2; ++j)
        acc[i][j] = __builtin_amdgcn_wmma_f32_16x16x32_f16(
            false, af[i], false, bf[j], (short)0, acc[i][j], false, false);
  }

  int lane = threadIdx.x & 31;
  int half = lane >> 4;
  int ncol = lane & 15;
  if constexpr (TR_STORE) {
    // C[n*ldc + m]; per lane fixed n, 8 consecutive m -> two b128 stores.
#pragma unroll
    for (int i = 0; i < 4; ++i)
#pragma unroll
      for (int j = 0; j < 2; ++j) {
        int n = n0 + 16 * j + ncol;
        int mbase = m0 + 16 * i + half * 8;
        float* p = C + (long)n * ldc + mbase;
        *(float4*)(p)     = make_float4(acc[i][j][0], acc[i][j][1],
                                        acc[i][j][2], acc[i][j][3]);
        *(float4*)(p + 4) = make_float4(acc[i][j][4], acc[i][j][5],
                                        acc[i][j][6], acc[i][j][7]);
      }
  } else {
#pragma unroll
    for (int i = 0; i < 4; ++i)
#pragma unroll
      for (int j = 0; j < 2; ++j) {
        int n = n0 + 16 * j + ncol;
        float bv = bias ? bias[n] : 0.f;
#pragma unroll
        for (int e = 0; e < 8; ++e) {
          int m = m0 + 16 * i + e + half * 8;
          float v = acc[i][j][e] + bv;
          if (relu) v = fmaxf(v, 0.f);
          C[(long)m * ldc + n] = v;
        }
      }
  }
}

// ---------------------------------------------------------------------------
// Stage 1 of attention, computed TRANSPOSED for contiguous f16 stores:
//   D[k',i] = sum_h v[k',b,h] * Qj[i,h],  Qj[i,h] = cq_rw*k[j,b,h] + cq_rr*rk[j,h]
// WMMA: A = V rows (k' x h), B = Qj cols (h x i).  D lane<->i (full_h row),
// elements<->k' (contiguous within full_h row) -> one b128 store per acc tile.
// Output f16 full_h[(b*256+i)][j*256+k'].
// Block: 256 threads = 8 waves; wave owns 32 i-rows, loops 4 x 64-k' tiles.
// ---------------------------------------------------------------------------
__global__ __launch_bounds__(256, 1) void
k_full(const float* __restrict__ heads,
       const float* __restrict__ rk,
       const float* __restrict__ cq_rw,
       const float* __restrict__ cq_rr,
       _Float16* __restrict__ full_h) {
  int b = blockIdx.x >> 8;
  int j = blockIdx.x & 255;
  int wv = threadIdx.x >> 5;
  int i0 = wv * 32;                 // this wave's i-rows (N dimension)
  int lane = threadIdx.x & 31;
  int half = lane >> 4;
  int lidx = lane & 15;

  const float* kvec = heads + (long)(j * kB + b) * (3 * kH) + kH;   // k[j,b,:]
  const float* rvec = rk + (long)j * kH;                            // rk[j,:]

  for (int kp0 = 0; kp0 < kS; kp0 += 64) {   // k' tiles (M dimension)
    v8f acc[4][2];                            // [k' tile][i tile]
#pragma unroll
    for (int mi = 0; mi < 4; ++mi)
#pragma unroll
      for (int nj = 0; nj < 2; ++nj) acc[mi][nj] = {};

    for (int h0 = 0; h0 < kH; h0 += 32) {
      // Row-invariant k / rk runs for this k-step (fragment element order).
      float kv[16], rv[16];
      load16f(kvec + h0 + half * 8, kv);
      load16f(rvec + h0 + half * 8, rv);

      // B operand: Qj columns (i), built fused in registers.
      v16h qf[2];
#pragma unroll
      for (int nj = 0; nj < 2; ++nj) {
        int row = i0 + 16 * nj + lidx;  // query index i
        float aw[16], ar[16];
        load16f(cq_rw + (long)(row * kB + b) * kH + h0 + half * 8, aw);
        load16f(cq_rr + (long)(row * kB + b) * kH + h0 + half * 8, ar);
#pragma unroll
        for (int e = 0; e < 16; ++e)
          qf[nj][e] = (_Float16)(aw[e] * kv[e] + ar[e] * rv[e]);
      }
      // A operand: V rows (k').
      v16h vf[4];
#pragma unroll
      for (int mi = 0; mi < 4; ++mi) {
        int col = kp0 + 16 * mi + lidx;  // key index k'
        float vv[16];
        load16f(heads + (long)(col * kB + b) * (3 * kH) + 2 * kH + h0 + half * 8, vv);
#pragma unroll
        for (int e = 0; e < 16; ++e) vf[mi][e] = (_Float16)vv[e];
      }
#pragma unroll
      for (int mi = 0; mi < 4; ++mi)
#pragma unroll
        for (int nj = 0; nj < 2; ++nj)
          acc[mi][nj] = __builtin_amdgcn_wmma_f32_16x16x32_f16(
              false, vf[mi], false, qf[nj], (short)0, acc[mi][nj], false, false);
    }
    // Store: per lane fixed i, 8 consecutive k' -> one 16B store per tile.
#pragma unroll
    for (int mi = 0; mi < 4; ++mi)
#pragma unroll
      for (int nj = 0; nj < 2; ++nj) {
        int i = i0 + 16 * nj + lidx;
        int kpbase = kp0 + 16 * mi + half * 8;
        v8h hv;
#pragma unroll
        for (int e = 0; e < 8; ++e) hv[e] = (_Float16)acc[mi][nj][e];
        *(v8h*)(full_h + (long)(b * kS + i) * 65536 + (long)j * kS + kpbase) = hv;
      }
  }
}

// ---------------------------------------------------------------------------
// Small elementwise kernels
// ---------------------------------------------------------------------------
__global__ void k_embed(const int* __restrict__ tokens,
                        const float* __restrict__ emb, float* __restrict__ w) {
  int r = blockIdx.x;      // token row s*B+b
  int d = threadIdx.x;     // 512 threads
  int t = tokens[r];
  w[(long)r * kDM + d] = emb[(long)t * kDM + d] * 22.62741699796952f;  // sqrt(512)
}

__global__ void k_posemb(float* __restrict__ r) {
  int s = blockIdx.x;      // 256
  int i = threadIdx.x;     // 256
  float pos = (float)(kS - 1 - s);
  float inv = __powf(10000.f, -(2.f * (float)i) / (float)kDM);
  float ang = pos * inv;
  r[(long)s * kDM + i] = __sinf(ang);
  r[(long)s * kDM + kDM / 2 + i] = __cosf(ang);
}

__global__ void k_cq(const float* __restrict__ heads,
                     const float* __restrict__ core,  // core_value[l] : [2][H]
                     const float* __restrict__ rwb, const float* __restrict__ rrb,
                     float* __restrict__ cq_rw, float* __restrict__ cq_rr) {
  int r = blockIdx.x;      // 512 token rows
  int h = threadIdx.x;     // 512
  float q = heads[(long)r * (3 * kH) + h];
  float c = 0.5f * (core[h] + core[kH + h]);   // mean over core_nums=2
  cq_rw[(long)r * kH + h] = (q + rwb[h]) * c;
  cq_rr[(long)r * kH + h] = (q + rrb[h]) * c;
}

// Residual + LayerNorm. Delta comes either from a dense [512x512] buffer, or
// from nchunks split-K partials laid out [c][bi][d] (bi = b*256 + s).
__global__ void k_residual_ln(const float* __restrict__ w,
                              const float* __restrict__ dense,
                              const float* __restrict__ part, int nchunks,
                              const float* __restrict__ gs,
                              const float* __restrict__ gb,
                              float* __restrict__ out) {
  __shared__ float red[256];
  __shared__ float red2[256];
  int r = blockIdx.x;          // token row s*B+b
  int tid = threadIdx.x;       // 256
  int s_tok = r >> 1, b = r & 1;
  int bi = b * kS + s_tok;

  float x0 = w[(long)r * kDM + tid];
  float x1 = w[(long)r * kDM + tid + 256];
  if (part) {
    const float* p = part + (long)bi * kDM;
    for (int c = 0; c < nchunks; ++c) {
      x0 += p[(long)c * kDM * 512 + tid];
      x1 += p[(long)c * kDM * 512 + tid + 256];
    }
  } else {
    x0 += dense[(long)r * kDM + tid];
    x1 += dense[(long)r * kDM + tid + 256];
  }
  red[tid]  = x0 + x1;
  red2[tid] = x0 * x0 + x1 * x1;
  __syncthreads();
  for (int st = 128; st > 0; st >>= 1) {
    if (tid < st) { red[tid] += red[tid + st]; red2[tid] += red2[tid + st]; }
    __syncthreads();
  }
  float mu  = red[0] * (1.f / kDM);
  float var = red2[0] * (1.f / kDM) - mu * mu;
  float inv = rsqrtf(var + kEPS);
  out[(long)r * kDM + tid]       = (x0 - mu) * inv * gs[tid] + gb[tid];
  out[(long)r * kDM + tid + 256] = (x1 - mu) * inv * gs[tid + 256] + gb[tid + 256];
}

// ---------------------------------------------------------------------------
extern "C" void kernel_launch(void* const* d_in, const int* in_sizes, int n_in,
                              void* d_out, int out_size, void* d_ws, size_t ws_size,
                              hipStream_t stream) {
  (void)in_sizes; (void)n_in; (void)out_size; (void)ws_size;
  const int*   tokens = (const int*)  d_in[0];
  const float* emb    = (const float*)d_in[1];
  const float* qkv_w  = (const float*)d_in[2];
  const float* r_w    = (const float*)d_in[3];
  const float* o_w    = (const float*)d_in[4];
  const float* core   = (const float*)d_in[5];
  const float* rwb    = (const float*)d_in[6];
  const float* rrb    = (const float*)d_in[7];
  const float* ln1_s  = (const float*)d_in[8];
  const float* ln1_b  = (const float*)d_in[9];
  const float* ff_w1  = (const float*)d_in[10];
  const float* ff_b1  = (const float*)d_in[11];
  const float* ff_w2  = (const float*)d_in[12];
  const float* ff_b2  = (const float*)d_in[13];
  const float* ln2_s  = (const float*)d_in[14];
  const float* ln2_b  = (const float*)d_in[15];

  char* ws = (char*)d_ws;
  size_t off = 0;
  auto alloc = [&](size_t bytes) {
    void* p = ws + off;
    off = (off + bytes + 255) & ~(size_t)255;
    return p;
  };
  float* w      = (float*)alloc((size_t)512 * 512 * 4);        // hidden [S*B, DM]
  float* rbuf   = (float*)alloc((size_t)256 * 512 * 4);        // pos emb [S, DM]
  float* headsb = (float*)alloc((size_t)512 * 1536 * 4);       // qkv [S*B, 3H]
  float* rkb    = (float*)alloc((size_t)256 * 512 * 4);        // rk [S, H]
  float* cq_rw  = (float*)alloc((size_t)512 * 512 * 4);
  float* cq_rr  = (float*)alloc((size_t)512 * 512 * 4);
  float* h1     = (float*)alloc((size_t)512 * 2048 * 4);       // FF hidden
  float* ffb    = (float*)alloc((size_t)512 * 512 * 4);        // FF out
  float* part   = (float*)alloc((size_t)16 * 512 * 512 * 4);   // split-K [c][bi][d]
  _Float16* fullh = (_Float16*)alloc((size_t)512 * 65536 * 2); // full [bi][j*256+k]

  k_embed <<<512, 512, 0, stream>>>(tokens, emb, w);
  k_posemb<<<256, 256, 0, stream>>>(rbuf);

  for (int l = 0; l < kL; ++l) {
    // heads = w @ qkv_w[l]^T : [512x512]x[512x1536]
    k_gemm_nt<false, false><<<dim3(4, 12, 1), 256, 0, stream>>>(
        w, kDM, qkv_w + (size_t)l * 3 * kH * kDM, nullptr, kDM,
        headsb, 3 * kH, 0, kDM, nullptr, 0);
    // rk = r @ r_w[l]^T : [256x512]x[512x512]
    k_gemm_nt<false, false><<<dim3(2, 4, 1), 256, 0, stream>>>(
        rbuf, kDM, r_w + (size_t)l * kH * kDM, nullptr, kDM,
        rkb, kH, 0, kDM, nullptr, 0);
    // cq_rw / cq_rr
    k_cq<<<512, 512, 0, stream>>>(headsb, core + (size_t)l * 2 * kH, rwb, rrb,
                                  cq_rw, cq_rr);
    // full[bi][j*256+k'] in f16 (512 WGs, one per (b,j))
    k_full<<<512, 256, 0, stream>>>(headsb, rkb, cq_rw, cq_rr, fullh);
    // attn: M=d (o_w rows), N=bi (full rows), K=65536 split into 16 chunks.
    // Transposed store -> part[c][bi][d], fully packed b128 stores.
    k_gemm_nt<true, true><<<dim3(4, 4, 16), 256, 0, stream>>>(
        o_w + (size_t)l * kDM * 65536, 65536, nullptr, fullh, 65536,
        part, 512, (long)512 * 512, 4096, nullptr, 0);
    // w = LN(w + attn_out)  (sums the 16 partials inside)
    k_residual_ln<<<512, 256, 0, stream>>>(w, nullptr, part, 16,
                                           ln1_s + (size_t)l * kDM,
                                           ln1_b + (size_t)l * kDM, w);
    // FF
    k_gemm_nt<false, false><<<dim3(4, 16, 1), 256, 0, stream>>>(
        w, kDM, ff_w1 + (size_t)l * kDI * kDM, nullptr, kDM,
        h1, kDI, 0, kDM, ff_b1 + (size_t)l * kDI, 1);
    k_gemm_nt<false, false><<<dim3(4, 4, 1), 256, 0, stream>>>(
        h1, kDI, ff_w2 + (size_t)l * kDM * kDI, nullptr, kDI,
        ffb, kDM, 0, kDI, ff_b2 + (size_t)l * kDM, 0);
    float* outp = (l == kL - 1) ? (float*)d_out : w;
    k_residual_ln<<<512, 256, 0, stream>>>(w, ffb, nullptr, 0,
                                           ln2_s + (size_t)l * kDM,
                                           ln2_b + (size_t)l * kDM, outp);
  }
}
